// FourierAttention_38912403701947
// MI455X (gfx1250) — matrix operand order repaired and checked by reading, not compiled, not run on hardware
//
#include <hip/hip_runtime.h>
#include <math.h>

typedef __attribute__((ext_vector_type(16))) _Float16 v16h;
typedef __attribute__((ext_vector_type(8)))  float    v8f;

#define NB 4
#define NL 4096
#define NH 16
#define NE 64
#define NG 25
#define SCALE 0.25f          // H^-0.5 = 16^-0.5

#define S1_CHUNKS 8
#define S1_CHUNK_L 512       // keys per block (L/8)
#define S1_WAVE_L  128       // keys per wave (4 waves/block)
#define REC_F 2112           // floats per partial record: 32 m + 32 s + 32*64 acc

__device__ __forceinline__ float rmax16(float v) {
  v = fmaxf(v, __shfl_xor(v, 1, 32));
  v = fmaxf(v, __shfl_xor(v, 2, 32));
  v = fmaxf(v, __shfl_xor(v, 4, 32));
  v = fmaxf(v, __shfl_xor(v, 8, 32));
  return v;
}
__device__ __forceinline__ float rsum16(float v) {
  v += __shfl_xor(v, 1, 32);
  v += __shfl_xor(v, 2, 32);
  v += __shfl_xor(v, 4, 32);
  v += __shfl_xor(v, 8, 32);
  return v;
}
__device__ __forceinline__ v8f wmma_f16(v16h a, v16h b, v8f c) {
  return __builtin_amdgcn_wmma_f32_16x16x32_f16(false, a, false, b, (short)0, c, false, false);
}

// ---------------- Stage 1: guidance tokens attend over keys (flash-style partials) ----
__global__ __launch_bounds__(128) void s1_partial(
    const float* __restrict__ K, const float* __restrict__ V,
    const float* __restrict__ GT, float* __restrict__ part)
{
  __shared__ _Float16 vt[4][NE][32];   // per-wave V^T tile (B-frag friendly)
  __shared__ _Float16 pl[4][16][32];   // per-wave P tile (C-layout -> A-layout bounce)
  __shared__ float accL[4][32][NE];
  __shared__ float mL[4][32];
  __shared__ float sL[4][32];
  __shared__ float cm[32], cs[32];

  const int tid  = threadIdx.x;
  const int lane = tid & 31;
  const int w    = tid >> 5;
  const int lo   = lane & 15;
  const int hi   = lane >> 4;

  const int bh    = blockIdx.x / S1_CHUNKS;
  const int chunk = blockIdx.x % S1_CHUNKS;
  const int b = bh / NH, h = bh % NH;

  // A-fragments of GT (M = 32 padded guidance rows, K = 64), pre-scaled
  v16h aGT[2][2];
  for (int mt = 0; mt < 2; ++mt) {
    const int g = mt * 16 + lo;
    const float* gr = (g < NG) ? (GT + (((size_t)b * NG + g) * NH + h) * NE) : nullptr;
    for (int ks = 0; ks < 2; ++ks) {
      v16h a;
      for (int j = 0; j < 8; ++j) {
        const int e0 = ks * 32 + hi * 8 + j;
        a[j]     = gr ? (_Float16)(gr[e0]      * SCALE) : (_Float16)0.f;
        a[j + 8] = gr ? (_Float16)(gr[e0 + 16] * SCALE) : (_Float16)0.f;
      }
      aGT[mt][ks] = a;
    }
  }

  const v8f zero = {0.f, 0.f, 0.f, 0.f, 0.f, 0.f, 0.f, 0.f};
  float m_run[2][8], s_run[2][8];
  v8f acc[2][4];
  for (int mt = 0; mt < 2; ++mt)
    for (int r = 0; r < 8; ++r) { m_run[mt][r] = -INFINITY; s_run[mt][r] = 0.f; }
  for (int mt = 0; mt < 2; ++mt)
    for (int et = 0; et < 4; ++et) acc[mt][et] = zero;

  const int l0 = chunk * S1_CHUNK_L + w * S1_WAVE_L;
  for (int step = 0; step < 4; ++step) {
    const int lb = l0 + step * 32;

    // stage 32 V rows into LDS transposed: vt[e][l_local]
    {
      const float* vr = V + (((size_t)b * NL + lb + lane) * NH + h) * NE;
      for (int e = 0; e < NE; ++e) vt[w][e][lane] = (_Float16)vr[e];
    }

    // scores S = (GT*scale) @ K^T for a 32x32 tile
    v8f sc[2][2];
    sc[0][0] = zero; sc[0][1] = zero; sc[1][0] = zero; sc[1][1] = zero;
    for (int nt = 0; nt < 2; ++nt) {
      const int l = lb + nt * 16 + lo;
      const float* kr = K + (((size_t)b * NL + l) * NH + h) * NE;
      for (int ks = 0; ks < 2; ++ks) {
        v16h bk;
        for (int j = 0; j < 16; ++j) bk[j] = (_Float16)kr[ks * 32 + hi * 8 + (j & 7) + ((j >> 3) << 4)];
        sc[0][nt] = wmma_f16(aGT[0][ks], bk, sc[0][nt]);
        sc[1][nt] = wmma_f16(aGT[1][ks], bk, sc[1][nt]);
      }
    }

    // online softmax over this 32-key tile + acc += P @ V
    for (int mt = 0; mt < 2; ++mt) {
      float p0[8], p1[8];
      for (int r = 0; r < 8; ++r) {
        const float a0 = sc[mt][0][r], a1 = sc[mt][1][r];
        const float tm = rmax16(fmaxf(a0, a1));
        const float nm = fmaxf(m_run[mt][r], tm);
        const float fac = __expf(m_run[mt][r] - nm);
        p0[r] = __expf(a0 - nm);
        p1[r] = __expf(a1 - nm);
        const float rs = rsum16(p0[r] + p1[r]);
        s_run[mt][r] = s_run[mt][r] * fac + rs;
        m_run[mt][r] = nm;
        for (int et = 0; et < 4; ++et) acc[mt][et][r] *= fac;
      }
      // bounce P through per-wave LDS: C-layout -> A-layout (16x32)
      for (int r = 0; r < 8; ++r) {
        const int m = r + hi * 8;
        pl[w][m][lo]      = (_Float16)p0[r];
        pl[w][m][16 + lo] = (_Float16)p1[r];
      }
      v16h pa;
      for (int j = 0; j < 8; ++j) {
        pa[j]     = pl[w][lo][hi * 8 + j];
        pa[j + 8] = pl[w][lo][hi * 8 + 16 + j];
      }
      for (int et = 0; et < 4; ++et) {
        v16h bv;
        for (int j = 0; j < 16; ++j) bv[j] = vt[w][et * 16 + lo][hi * 8 + (j & 7) + ((j >> 3) << 4)];
        acc[mt][et] = wmma_f16(pa, bv, acc[mt][et]);
      }
    }
  }

  // dump per-wave state, combine the 4 waves, emit one partial record
  for (int mt = 0; mt < 2; ++mt)
    for (int r = 0; r < 8; ++r) {
      const int g = mt * 16 + r + hi * 8;
      if (lo == 0) { mL[w][g] = m_run[mt][r]; sL[w][g] = s_run[mt][r]; }
      for (int et = 0; et < 4; ++et) accL[w][g][et * 16 + lo] = acc[mt][et][r];
    }
  __syncthreads();
  if (tid < 32) {
    const int g = tid;
    float M = mL[0][g];
    for (int ww = 1; ww < 4; ++ww) M = fmaxf(M, mL[ww][g]);
    float S = 0.f;
    for (int ww = 0; ww < 4; ++ww) S += sL[ww][g] * __expf(mL[ww][g] - M);
    cm[g] = M; cs[g] = S;
  }
  __syncthreads();
  float* rec = part + (size_t)blockIdx.x * REC_F;
  if (tid < 32) { rec[tid] = cm[tid]; rec[32 + tid] = cs[tid]; }
  for (int idx = tid; idx < 32 * NE; idx += 128) {
    const int g = idx >> 6;
    float a = 0.f;
    for (int ww = 0; ww < 4; ++ww) a += accL[ww][g][idx & 63] * __expf(mL[ww][g] - cm[g]);
    rec[64 + idx] = a;
  }
}

// ---------------- Stage 1b: combine the 8 L-chunks per (b,h) -> guidance_v ------------
__global__ __launch_bounds__(128) void s1_reduce(const float* __restrict__ part,
                                                 float* __restrict__ gv)
{
  __shared__ float cm[32], cs[32];
  const int tid = threadIdx.x;
  const int bh  = blockIdx.x;
  const float* base = part + (size_t)bh * S1_CHUNKS * REC_F;
  if (tid < 32) {
    float M = -INFINITY;
    for (int c = 0; c < S1_CHUNKS; ++c) M = fmaxf(M, base[c * REC_F + tid]);
    float S = 0.f;
    for (int c = 0; c < S1_CHUNKS; ++c)
      S += base[c * REC_F + 32 + tid] * __expf(base[c * REC_F + tid] - M);
    cm[tid] = M; cs[tid] = (S > 0.f) ? S : 1.f;
  }
  __syncthreads();
  for (int idx = tid; idx < 32 * NE; idx += 128) {
    const int g = idx >> 6;
    float a = 0.f;
    for (int c = 0; c < S1_CHUNKS; ++c)
      a += base[c * REC_F + 64 + idx] * __expf(base[c * REC_F + g] - cm[g]);
    gv[(size_t)bh * 32 * NE + idx] = a / cs[g];
  }
}

// ---------------- Stage 2: queries attend over the 25 guidance tokens ----------------
__global__ __launch_bounds__(128) void s2_attend(
    const float* __restrict__ Q, const float* __restrict__ GT,
    const float* __restrict__ gv, float* __restrict__ out)
{
  __shared__ _Float16 gtB[32][NE];   // [g][e] (rows 25..31 zeroed)
  __shared__ _Float16 gvT[NE][32];   // [e][g]
  __shared__ _Float16 pl[4][16][32];

  const int tid  = threadIdx.x;
  const int lane = tid & 31;
  const int w    = tid >> 5;
  const int lo   = lane & 15;
  const int hi   = lane >> 4;
  const int bh = blockIdx.x >> 5;
  const int qc = blockIdx.x & 31;
  const int b = bh >> 4, h = bh & 15;

  for (int idx = tid; idx < 32 * NE; idx += 128) {
    const int g = idx >> 6, e = idx & 63;
    gtB[g][e] = (_Float16)((g < NG) ? GT[(((size_t)b * NG + g) * NH + h) * NE + e] : 0.f);
    const int e2 = idx >> 5, g2 = idx & 31;
    gvT[e2][g2] = (_Float16)((g2 < NG) ? gv[(size_t)bh * 32 * NE + g2 * NE + e2] : 0.f);
  }
  __syncthreads();

  const v8f zero = {0.f, 0.f, 0.f, 0.f, 0.f, 0.f, 0.f, 0.f};
  for (int mt = 0; mt < 2; ++mt) {
    const int lbase = qc * 128 + w * 32 + mt * 16;
    const float* qr = Q + (((size_t)b * NL + lbase + lo) * NH + h) * NE;
    v8f s0 = zero, s1 = zero;
    for (int ks = 0; ks < 2; ++ks) {
      v16h aq;
      for (int j = 0; j < 8; ++j) {
        const int e0 = ks * 32 + hi * 8 + j;
        aq[j]     = (_Float16)(qr[e0]      * SCALE);
        aq[j + 8] = (_Float16)(qr[e0 + 16] * SCALE);
      }
      v16h b0, b1;
      for (int j = 0; j < 16; ++j) {
        b0[j] = gtB[lo][ks * 32 + hi * 8 + (j & 7) + ((j >> 3) << 4)];
        b1[j] = gtB[16 + lo][ks * 32 + hi * 8 + (j & 7) + ((j >> 3) << 4)];
      }
      s0 = wmma_f16(aq, b0, s0);
      s1 = wmma_f16(aq, b1, s1);
    }
    const bool v1ok = (16 + lo) < NG;   // mask padded guidance columns
    for (int r = 0; r < 8; ++r) {
      const float a0 = s0[r];
      const float a1 = v1ok ? s1[r] : -INFINITY;
      const float mx = rmax16(fmaxf(a0, a1));
      const float p0 = __expf(a0 - mx);
      const float p1 = v1ok ? __expf(s1[r] - mx) : 0.f;
      const float inv = 1.f / rsum16(p0 + p1);
      const int m = r + hi * 8;
      pl[w][m][lo]      = (_Float16)(p0 * inv);
      pl[w][m][16 + lo] = (_Float16)(p1 * inv);
    }
    v16h pa;
    for (int j = 0; j < 8; ++j) {
      pa[j]     = pl[w][lo][hi * 8 + j];
      pa[j + 8] = pl[w][lo][hi * 8 + 16 + j];
    }
    for (int et = 0; et < 4; ++et) {
      v16h bv;
      for (int j = 0; j < 16; ++j) bv[j] = gvT[et * 16 + lo][hi * 8 + (j & 7) + ((j >> 3) << 4)];
      v8f o = wmma_f16(pa, bv, zero);
      for (int r = 0; r < 8; ++r) {
        const int ll = lbase + r + hi * 8;
        out[(((size_t)b * NL + ll) * NH + h) * NE + et * 16 + lo] = o[r];
      }
    }
  }
}

extern "C" void kernel_launch(void* const* d_in, const int* in_sizes, int n_in,
                              void* d_out, int out_size, void* d_ws, size_t ws_size,
                              hipStream_t stream) {
  const float* Q  = (const float*)d_in[0];
  const float* K  = (const float*)d_in[1];
  const float* V  = (const float*)d_in[2];
  const float* GT = (const float*)d_in[3];
  float* out  = (float*)d_out;
  float* part = (float*)d_ws;                              // 512 * 2112 floats
  float* gv   = part + (size_t)NB * NH * S1_CHUNKS * REC_F; // 64 * 32 * 64 floats

  s1_partial<<<NB * NH * S1_CHUNKS, 128, 0, stream>>>(K, V, GT, part);
  s1_reduce <<<NB * NH,             128, 0, stream>>>(part, gv);
  s2_attend <<<NB * NH * 32,        128, 0, stream>>>(Q, GT, gv, out);
}
